// AlmondGPTModel_9783935500757
// MI455X (gfx1250) — matrix operand order repaired and checked by reading, not compiled, not run on hardware
//
#include <hip/hip_runtime.h>
#include <stdint.h>
#include <stddef.h>

// Problem constants (match reference)
#define Bz  2
#define Tz  1024
#define Dz  1024
#define Hz  16
#define HDz 64
#define Lz  8
#define FFz 4096
#define Vz  32000
#define Mz  (Bz * Tz)   // 2048 rows

#define USE_TDM 1

typedef __bf16 bf16;
typedef __attribute__((ext_vector_type(16))) __bf16 bf16x16;
typedef __attribute__((ext_vector_type(8)))  __bf16 bf16x8;
typedef __attribute__((ext_vector_type(8)))  float  f32x8;
typedef __attribute__((ext_vector_type(4)))  uint32_t u32x4;
typedef __attribute__((ext_vector_type(8)))  uint32_t u32x8;

union bfrag { bf16x16 v; bf16x8 h2[2]; };

// f32 -> bf16 round-to-nearest-even (bit-level)
static __device__ __forceinline__ bf16 f2bf(float f) {
  union { float f; uint32_t u; } c; c.f = f;
  uint32_t r = (c.u + 0x7FFFu + ((c.u >> 16) & 1u)) >> 16;
  uint16_t lo = (uint16_t)r;
  return __builtin_bit_cast(bf16, lo);
}

#if USE_TDM
// Issue one Tensor-Data-Mover DMA: 64 rows x 32 cols of bf16 from global
// (row stride = strideElems) into LDS at lds_off, contiguous row-major.
// D# per CDNA5 ISA (2D tile -> groups 0 and 1 only).
static __device__ __forceinline__ void tdm_load_tile64x32(const bf16* g, uint32_t lds_off,
                                                          uint32_t strideElems) {
  uint64_t ga  = (uint64_t)(uintptr_t)g;
  uint32_t glo = (uint32_t)__builtin_amdgcn_readfirstlane((int)(uint32_t)ga);
  uint32_t ghi = (uint32_t)__builtin_amdgcn_readfirstlane((int)(uint32_t)(ga >> 32));
  uint32_t lof = (uint32_t)__builtin_amdgcn_readfirstlane((int)lds_off);
  uint32_t str = (uint32_t)__builtin_amdgcn_readfirstlane((int)strideElems);
  u32x4 g0;
  g0[0] = 1u;                                   // count=1 (valid user descriptor)
  g0[1] = lof;                                  // lds_addr (bytes)
  g0[2] = glo;                                  // global_addr[31:0]
  g0[3] = (ghi & 0x01FFFFFFu) | (2u << 30);     // global_addr[56:32] | type=2
  u32x8 g1;
  g1[0] = 1u << 16;                             // data_size=1 -> 2 bytes/elem
  g1[1] = 32u << 16;                            // tensor_dim0 = 32
  g1[2] = 64u << 16;                            // tensor_dim1 = 64
  g1[3] = 32u << 16;                            // tile_dim0 = 32
  g1[4] = 64u;                                  // tile_dim1 = 64, tile_dim2 = 0
  g1[5] = str;                                  // tensor_dim0_stride (lo 32)
  g1[6] = 0u;
  g1[7] = 0u;
  asm volatile("tensor_load_to_lds %0, %1" :: "s"(g0), "s"(g1) : "memory");
}
#endif

// ---------------------------------------------------------------- transpose + convert
// src[Ksrc][Nsrc] f32  ->  dst[Nsrc][Ksrc] bf16   (tiled, coalesced both ways)
__global__ __launch_bounds__(256)
void transpose_f32_to_bf16_kernel(const float* __restrict__ src, bf16* __restrict__ dst,
                                  int Ksrc, int Nsrc) {
  __shared__ float tile[32][33];
  int n0 = blockIdx.x * 32, k0 = blockIdx.y * 32;
  int tx = threadIdx.x & 31, ty = threadIdx.x >> 5;   // 32 x 8
  for (int j = ty; j < 32; j += 8)
    tile[j][tx] = src[(size_t)(k0 + j) * Nsrc + n0 + tx];
  __syncthreads();
  for (int j = ty; j < 32; j += 8)
    dst[(size_t)(n0 + j) * Ksrc + k0 + tx] = f2bf(tile[tx][j]);
}

// ---------------------------------------------------------------- embedding
__global__ __launch_bounds__(256)
void embed_kernel(const int* __restrict__ x, const float* __restrict__ tok,
                  const float* __restrict__ pos, float* __restrict__ h) {
  int row = blockIdx.x;            // 0..Mz-1
  int t   = row % Tz;
  int tk  = x[row];
  const float* te = tok + (size_t)tk * Dz;
  const float* pe = pos + (size_t)t * Dz;
  float* hr = h + (size_t)row * Dz;
  for (int i = threadIdx.x; i < Dz; i += 256) hr[i] = te[i] + pe[i];
}

// ---------------------------------------------------------------- layernorm (f32 in, bf16 out)
__global__ __launch_bounds__(256)
void layernorm_kernel(const float* __restrict__ X, const float* __restrict__ g,
                      const float* __restrict__ b, bf16* __restrict__ out) {
  __shared__ float red[256];
  int row = blockIdx.x, tid = threadIdx.x;
  const float* x = X + (size_t)row * Dz;
  float s = 0.f;
  for (int i = tid; i < Dz; i += 256) s += x[i];
  red[tid] = s; __syncthreads();
  for (int o = 128; o > 0; o >>= 1) { if (tid < o) red[tid] += red[tid + o]; __syncthreads(); }
  float mean = red[0] * (1.f / Dz);
  __syncthreads();
  float v = 0.f;
  for (int i = tid; i < Dz; i += 256) { float d = x[i] - mean; v += d * d; }
  red[tid] = v; __syncthreads();
  for (int o = 128; o > 0; o >>= 1) { if (tid < o) red[tid] += red[tid + o]; __syncthreads(); }
  float inv = rsqrtf(red[0] * (1.f / Dz) + 1e-5f);
  bf16* orow = out + (size_t)row * Dz;
  for (int i = tid; i < Dz; i += 256) orow[i] = f2bf((x[i] - mean) * inv * g[i] + b[i]);
}

// ---------------------------------------------------------------- WMMA GEMM
// C[M,N] = A[M,K](bf16 rowmajor) @ W (given as Wt[N,K] bf16 rowmajor)
// 64x64 block tile, 8 waves (4 m x 2 n), each wave 16x32 output.
// Double-buffered LDS tiles streamed by the Tensor Data Mover (wave 0 issues,
// s_wait_tensorcnt + workgroup barrier publishes). Both tiles are contiguous
// 64x32 row-major copies, so TDM handles them without any lane staging code.
template<int BIAS, int RELU, int RESID, int OUTB, int OUTT>
__global__ __launch_bounds__(256)
void gemm_kernel(const bf16* __restrict__ A, const bf16* __restrict__ Wt,
                 const float* __restrict__ bias, const float* __restrict__ resid,
                 float* __restrict__ outf, bf16* __restrict__ outb,
                 int N, int K) {
  __shared__ __align__(16) bf16 sA[2][64 * 32];   // [m][k]
  __shared__ __align__(16) bf16 sB[2][64 * 32];   // [n][k]
  int tid  = threadIdx.x;
  int lane = tid & 31, wave = tid >> 5;
  int half = lane >> 4, l16 = lane & 15;
  int wm = (wave & 3) * 16, wn = (wave >> 2) * 32;
  int mBlk = blockIdx.y * 64, nBlk = blockIdx.x * 64;
  const bf16* aBase = A  + (size_t)mBlk * K;
  const bf16* bBase = Wt + (size_t)nBlk * K;

  f32x8 acc0 = {}, acc1 = {};
  int nk = K >> 5;

#if USE_TDM
  bool issuer = (wave == 0);
  if (issuer) {
    tdm_load_tile64x32(aBase, (uint32_t)(size_t)&sA[0][0], (uint32_t)K);
    tdm_load_tile64x32(bBase, (uint32_t)(size_t)&sB[0][0], (uint32_t)K);
    __builtin_amdgcn_s_wait_tensorcnt(0);
  }
  __syncthreads();
#endif

  for (int kt = 0; kt < nk; ++kt) {
#if USE_TDM
    int cur = kt & 1;
    if (issuer && (kt + 1) < nk) {   // stream next tile while computing this one
      tdm_load_tile64x32(aBase + (kt + 1) * 32, (uint32_t)(size_t)&sA[cur ^ 1][0], (uint32_t)K);
      tdm_load_tile64x32(bBase + (kt + 1) * 32, (uint32_t)(size_t)&sB[cur ^ 1][0], (uint32_t)K);
    }
#else
    int cur = 0;
    {
      int r = tid >> 2, c = (tid & 3) * 8;
      *(bf16x8*)(&sA[0][r * 32 + c]) = *(const bf16x8*)(aBase + (size_t)r * K + kt * 32 + c);
      *(bf16x8*)(&sB[0][r * 32 + c]) = *(const bf16x8*)(bBase + (size_t)r * K + kt * 32 + c);
    }
    __syncthreads();
#endif

    // A fragment: lane m = l16; e<8 -> K=half*8+e ; e>=8 -> K=16+half*8+(e-8)
    bfrag af, b0, b1;
    const bf16* pa = &sA[cur][(wm + l16) * 32];
    af.h2[0] = *(const bf16x8*)(pa + half * 8);
    af.h2[1] = *(const bf16x8*)(pa + 16 + half * 8);
    // B fragments: lane n = l16; e<8 -> K=half*8+e ; e>=8 -> K=16+half*8+(e-8)
    const bf16* pb0 = &sB[cur][(wn + l16) * 32];
    const bf16* pb1 = pb0 + 16 * 32;
    b0.h2[0] = *(const bf16x8*)(pb0 + half * 8);
    b0.h2[1] = *(const bf16x8*)(pb0 + 16 + half * 8);
    b1.h2[0] = *(const bf16x8*)(pb1 + half * 8);
    b1.h2[1] = *(const bf16x8*)(pb1 + 16 + half * 8);

    acc0 = __builtin_amdgcn_wmma_f32_16x16x32_bf16(false, af.v, false, b0.v,
                                                   (short)0, acc0, false, false);
    acc1 = __builtin_amdgcn_wmma_f32_16x16x32_bf16(false, af.v, false, b1.v,
                                                   (short)0, acc1, false, false);

#if USE_TDM
    if (issuer) __builtin_amdgcn_s_wait_tensorcnt(0);  // next tile landed
    __syncthreads();
#else
    __syncthreads();
#endif
  }

  // C/D layout: VGPR r -> M = r + 8*half, N = l16
  int row0 = mBlk + wm + half * 8;
  int col0 = nBlk + wn + l16, col1 = col0 + 16;
  float bi0 = BIAS ? bias[col0] : 0.f;
  float bi1 = BIAS ? bias[col1] : 0.f;
  #pragma unroll
  for (int r = 0; r < 8; ++r) {
    float v0 = acc0[r] + bi0, v1 = acc1[r] + bi1;
    if (RELU) { v0 = fmaxf(v0, 0.f); v1 = fmaxf(v1, 0.f); }
    if (OUTT) {
      // transposed bf16 store: out[col][row], row stride = Mz (consecutive in r)
      outb[(size_t)col0 * Mz + row0 + r] = f2bf(v0);
      outb[(size_t)col1 * Mz + row0 + r] = f2bf(v1);
    } else {
      size_t o0 = (size_t)(row0 + r) * N + col0;
      size_t o1 = (size_t)(row0 + r) * N + col1;
      if (RESID) { v0 += resid[o0]; v1 += resid[o1]; }
      if (OUTB) { outb[o0] = f2bf(v0); outb[o1] = f2bf(v1); }
      else      { outf[o0] = v0;       outf[o1] = v1; }
    }
  }
}

// ---------------------------------------------------------------- flash attention
// One wave per (b, h, 16-query tile). Causal, online softmax. QK^T and PV via
// WMMA bf16. V is consumed pre-transposed (Vt[D][B*T]) so PV B-fragments are
// contiguous global reads; only P needs a 1KB LDS re-layout (C- to A-layout).
__global__ __launch_bounds__(32)
void attn_kernel(const bf16* __restrict__ Q, const bf16* __restrict__ Kb,
                 const bf16* __restrict__ Vt, bf16* __restrict__ O) {
  __shared__ __align__(16) bf16 sP[16 * 32];     // P tile [q][key]
  const float scale = 0.125f;                    // 1/sqrt(HD=64)
  int lane = threadIdx.x;
  int half = lane >> 4, l16 = lane & 15;
  int q0 = blockIdx.x * 16;
  int b = blockIdx.y / Hz, h = blockIdx.y % Hz;
  const bf16* qhead = Q  + (size_t)b * Tz * Dz + h * HDz;
  const bf16* khead = Kb + (size_t)b * Tz * Dz + h * HDz;

  // Q fragments (A-layout): 2 chunks of 32 over HD=64
  bfrag qf[2];
  #pragma unroll
  for (int c = 0; c < 2; ++c) {
    const bf16* qr = qhead + (size_t)(q0 + l16) * Dz + c * 32 + half * 8;
    qf[c].h2[0] = *(const bf16x8*)qr;
    qf[c].h2[1] = *(const bf16x8*)(qr + 16);
  }

  f32x8 oacc[4] = {{}, {}, {}, {}};
  float mrow[8], lrow[8];
  #pragma unroll
  for (int r = 0; r < 8; ++r) { mrow[r] = -3e38f; lrow[r] = 0.f; }

  int nkb = q0 / 32 + 1;   // causal key blocks of 32 covering keys <= q0+15
  for (int kb = 0; kb < nkb; ++kb) {
    int kbase = kb * 32;
    // S = Q K^T : two 16x16 C tiles
    f32x8 sacc[2] = {{}, {}};
    #pragma unroll
    for (int tile = 0; tile < 2; ++tile) {
      int key = kbase + tile * 16 + l16;      // B-frag: n = key column
      const bf16* krow = khead + (size_t)key * Dz;
      #pragma unroll
      for (int c = 0; c < 2; ++c) {
        bfrag kf;                              // e<8 -> d = c*32 + half*8 + e ; e>=8 -> d = c*32 + 16 + half*8 + (e-8)
        const bf16* kr = krow + c * 32 + half * 8;
        kf.h2[0] = *(const bf16x8*)kr;
        kf.h2[1] = *(const bf16x8*)(kr + 16);
        sacc[tile] = __builtin_amdgcn_wmma_f32_16x16x32_bf16(
            false, qf[c].v, false, kf.v, (short)0, sacc[tile], false, false);
      }
    }

    // online softmax over this 32-key block
    float p0[8], p1[8], alph[8];
    #pragma unroll
    for (int r = 0; r < 8; ++r) {
      int qg  = q0 + r + 8 * half;             // C-layout row
      int kg0 = kbase + l16, kg1 = kbase + 16 + l16;
      float v0 = (kg0 <= qg) ? sacc[0][r] * scale : -3e38f;
      float v1 = (kg1 <= qg) ? sacc[1][r] * scale : -3e38f;
      float mx = fmaxf(v0, v1);
      #pragma unroll
      for (int o = 1; o < 16; o <<= 1) mx = fmaxf(mx, __shfl_xor(mx, o, 32));
      float mn = fmaxf(mrow[r], mx);
      float a  = __expf(mrow[r] - mn);
      v0 = __expf(v0 - mn); v1 = __expf(v1 - mn);
      float rs = v0 + v1;
      #pragma unroll
      for (int o = 1; o < 16; o <<= 1) rs += __shfl_xor(rs, o, 32);
      lrow[r] = lrow[r] * a + rs;
      mrow[r] = mn; alph[r] = a; p0[r] = v0; p1[r] = v1;
    }
    #pragma unroll
    for (int nt = 0; nt < 4; ++nt)
      #pragma unroll
      for (int r = 0; r < 8; ++r) oacc[nt][r] *= alph[r];

    // write P (C-layout) to LDS row-major [q][key]
    #pragma unroll
    for (int r = 0; r < 8; ++r) {
      int prow = r + 8 * half;
      sP[prow * 32 + l16]      = f2bf(p0[r]);
      sP[prow * 32 + 16 + l16] = f2bf(p1[r]);
    }
    __syncthreads();

    // P as A-fragment; Vt columns as B-fragments (contiguous global reads)
    bfrag pf;
    pf.h2[0] = *(const bf16x8*)(sP + l16 * 32 + half * 8);
    pf.h2[1] = *(const bf16x8*)(sP + l16 * 32 + 16 + half * 8);
    #pragma unroll
    for (int nt = 0; nt < 4; ++nt) {
      bfrag vf;  // e<8 -> key = kbase + half*8 + e ; e>=8 -> key = kbase + 16 + half*8 + (e-8) ; d = h*64 + nt*16 + l16
      const bf16* vc = Vt + (size_t)(h * HDz + nt * 16 + l16) * Mz
                          + (size_t)b * Tz + kbase + half * 8;
      vf.h2[0] = *(const bf16x8*)vc;
      vf.h2[1] = *(const bf16x8*)(vc + 16);
      oacc[nt] = __builtin_amdgcn_wmma_f32_16x16x32_bf16(
          false, pf.v, false, vf.v, (short)0, oacc[nt], false, false);
    }
    __syncthreads();
  }

  // normalize and store O (bf16, [B*T, D] with head offset)
  #pragma unroll
  for (int r = 0; r < 8; ++r) {
    float invl = 1.f / lrow[r];
    int row = q0 + r + 8 * half;
    bf16* orow = O + (size_t)(b * Tz + row) * Dz + h * HDz;
    #pragma unroll
    for (int nt = 0; nt < 4; ++nt)
      orow[nt * 16 + l16] = f2bf(oacc[nt][r] * invl);
  }
}

// ---------------------------------------------------------------- loss
__global__ __launch_bounds__(256)
void rowloss_kernel(const float* __restrict__ logits, const int* __restrict__ tgt,
                    float* __restrict__ rowloss) {
  __shared__ float red[256];
  int row = blockIdx.x, tid = threadIdx.x;
  const float* lg = logits + (size_t)row * Vz;
  float mx = -3e38f;
  for (int i = tid; i < Vz; i += 256) mx = fmaxf(mx, lg[i]);
  red[tid] = mx; __syncthreads();
  for (int o = 128; o > 0; o >>= 1) { if (tid < o) red[tid] = fmaxf(red[tid], red[tid + o]); __syncthreads(); }
  float m = red[0]; __syncthreads();
  float s = 0.f;
  for (int i = tid; i < Vz; i += 256) s += __expf(lg[i] - m);
  red[tid] = s; __syncthreads();
  for (int o = 128; o > 0; o >>= 1) { if (tid < o) red[tid] += red[tid + o]; __syncthreads(); }
  if (tid == 0) rowloss[row] = (m + __logf(red[0])) - lg[tgt[row]];
}

__global__ __launch_bounds__(256)
void loss_final_kernel(const float* __restrict__ rowloss, float* __restrict__ out) {
  __shared__ float red[256];
  int tid = threadIdx.x;
  float s = 0.f;
  for (int i = tid; i < Mz; i += 256) s += rowloss[i];
  red[tid] = s; __syncthreads();
  for (int o = 128; o > 0; o >>= 1) { if (tid < o) red[tid] += red[tid + o]; __syncthreads(); }
  if (tid == 0) out[0] = red[0] * (1.f / Mz);
}

// ---------------------------------------------------------------- host launcher
extern "C" void kernel_launch(void* const* d_in, const int* in_sizes, int n_in,
                              void* d_out, int out_size, void* d_ws, size_t ws_size,
                              hipStream_t stream) {
  (void)in_sizes; (void)n_in; (void)out_size; (void)ws_size;

  const int*   x       = (const int*)  d_in[0];
  const int*   targets = (const int*)  d_in[1];
  const float* tok_emb = (const float*)d_in[2];
  const float* pos_emb = (const float*)d_in[3];
  const float* Wq      = (const float*)d_in[4];
  const float* Wk      = (const float*)d_in[5];
  const float* Wv      = (const float*)d_in[6];
  const float* Wo      = (const float*)d_in[7];
  const float* bo      = (const float*)d_in[8];
  const float* W1      = (const float*)d_in[9];
  const float* b1      = (const float*)d_in[10];
  const float* W2      = (const float*)d_in[11];
  const float* b2      = (const float*)d_in[12];
  const float* ln1_g   = (const float*)d_in[13];
  const float* ln1_b   = (const float*)d_in[14];
  const float* ln2_g   = (const float*)d_in[15];
  const float* ln2_b   = (const float*)d_in[16];
  const float* lnf_g   = (const float*)d_in[17];
  const float* lnf_b   = (const float*)d_in[18];
  const float* lm_W    = (const float*)d_in[19];
  const float* lm_b    = (const float*)d_in[20];

  float* logits = (float*)d_out;
  float* loss   = logits + (size_t)Mz * Vz;

  // workspace carve-up (256B aligned)
  uint8_t* ws = (uint8_t*)d_ws;
  size_t off = 0;
  auto take = [&](size_t bytes) -> uint8_t* {
    uint8_t* p = ws + off;
    off = (off + bytes + 255) & ~(size_t)255;
    return p;
  };
  float* h      = (float*)take((size_t)Mz * Dz * 4);
  bf16*  xn     = (bf16*) take((size_t)Mz * Dz * 2);
  bf16*  qb     = (bf16*) take((size_t)Mz * Dz * 2);
  bf16*  kb     = (bf16*) take((size_t)Mz * Dz * 2);
  bf16*  vt     = (bf16*) take((size_t)Dz * Mz * 2);   // V transposed [D][B*T]
  bf16*  ob     = (bf16*) take((size_t)Mz * Dz * 2);
  bf16*  ff     = (bf16*) take((size_t)Mz * FFz * 2);
  bf16*  wl     = (bf16*) take(((size_t)4 * Dz * Dz + 2 * (size_t)Dz * FFz) * 2);
  bf16*  lmw    = (bf16*) take((size_t)Dz * Vz * 2);   // lm_W^T [V][D]
  float* rowls  = (float*)take((size_t)Mz * 4);

  // transpose-convert: src[K][N] f32 -> dst[N][K] bf16
  auto cvtT = [&](const float* s, bf16* d, int Ksrc, int Nsrc) {
    transpose_f32_to_bf16_kernel<<<dim3(Nsrc / 32, Ksrc / 32), 256, 0, stream>>>(s, d, Ksrc, Nsrc);
  };

  const size_t DD = (size_t)Dz * Dz;
  bf16* wq_t = wl;                                 // [D][D]
  bf16* wk_t = wl + DD;
  bf16* wv_t = wl + 2 * DD;
  bf16* wo_t = wl + 3 * DD;
  bf16* w1_t = wl + 4 * DD;                        // [FF][D]
  bf16* w2_t = wl + 4 * DD + (size_t)Dz * FFz;     // [D][FF]

  // LM head weight transposed once (resident): [D][V] -> [V][D]
  cvtT(lm_W, lmw, Dz, Vz);

  // embeddings
  embed_kernel<<<Mz, 256, 0, stream>>>(x, tok_emb, pos_emb, h);

  dim3 gD (Dz / 64,  Mz / 64);   // N=1024
  dim3 gFF(FFz / 64, Mz / 64);   // N=4096
  dim3 gV (Vz / 64,  Mz / 64);   // N=32000

  for (int l = 0; l < Lz; ++l) {
    // transpose-convert this layer's weights into bf16 scratch
    cvtT(Wq + (size_t)l * DD, wq_t, Dz, Dz);
    cvtT(Wk + (size_t)l * DD, wk_t, Dz, Dz);
    cvtT(Wv + (size_t)l * DD, wv_t, Dz, Dz);
    cvtT(Wo + (size_t)l * DD, wo_t, Dz, Dz);
    cvtT(W1 + (size_t)l * Dz * FFz, w1_t, Dz, FFz);
    cvtT(W2 + (size_t)l * FFz * Dz, w2_t, FFz, Dz);

    // LN1 -> xn (bf16)
    layernorm_kernel<<<Mz, 256, 0, stream>>>(h, ln1_g + (size_t)l * Dz,
                                             ln1_b + (size_t)l * Dz, xn);
    // QKV projections (bf16 out, no bias); V written transposed for attention
    gemm_kernel<0,0,0,1,0><<<gD, 256, 0, stream>>>(xn, wq_t, nullptr, nullptr, nullptr, qb, Dz, Dz);
    gemm_kernel<0,0,0,1,0><<<gD, 256, 0, stream>>>(xn, wk_t, nullptr, nullptr, nullptr, kb, Dz, Dz);
    gemm_kernel<0,0,0,1,1><<<gD, 256, 0, stream>>>(xn, wv_t, nullptr, nullptr, nullptr, vt, Dz, Dz);

    // causal flash attention
    attn_kernel<<<dim3(Tz / 16, Bz * Hz), 32, 0, stream>>>(qb, kb, vt, ob);

    // out projection + bias + residual (f32 h)
    gemm_kernel<1,0,1,0,0><<<gD, 256, 0, stream>>>(ob, wo_t, bo + (size_t)l * Dz, h, h, nullptr, Dz, Dz);

    // LN2 -> xn
    layernorm_kernel<<<Mz, 256, 0, stream>>>(h, ln2_g + (size_t)l * Dz,
                                             ln2_b + (size_t)l * Dz, xn);
    // FFN: relu(xn @ W1 + b1) -> ff (bf16)
    gemm_kernel<1,1,0,1,0><<<gFF, 256, 0, stream>>>(xn, w1_t, b1 + (size_t)l * FFz, nullptr, nullptr, ff, FFz, Dz);
    // ff @ W2 + b2 + residual -> h
    gemm_kernel<1,0,1,0,0><<<gD, 256, 0, stream>>>(ff, w2_t, b2 + (size_t)l * Dz, h, h, nullptr, Dz, FFz);
  }

  // final LN
  layernorm_kernel<<<Mz, 256, 0, stream>>>(h, lnf_g, lnf_b, xn);
  // LM head: logits = xn @ lm_W + lm_b  (f32, straight into d_out)
  gemm_kernel<1,0,0,0,0><<<gV, 256, 0, stream>>>(xn, lmw, lm_b, nullptr, logits, nullptr, Vz, Dz);

  // cross-entropy (deterministic two-stage reduction)
  rowloss_kernel<<<Mz, 256, 0, stream>>>(logits, targets, rowls);
  loss_final_kernel<<<1, 256, 0, stream>>>(rowls, loss);
}
